// GATLayer_56057913147782
// MI455X (gfx1250) — matrix operand order repaired and checked by reading, not compiled, not run on hardware
//
#include <hip/hip_runtime.h>

// GAT layer, fused flash-style implementation for gfx1250 (MI455X).
// N=8192, IN_F=512, OUT_F=64.
// adj is read exactly once with non-temporal hints (268 MB -> ~11.5 us floor at
// 23.3 TB/s); both GEMMs run on v_wmma_f32_16x16x32_bf16. Softmax uses a fixed
// per-row shift C_i = lrelu(Wh1[i] + max_j Wh2[j]) (exact by shift invariance,
// overflow-safe since it upper-bounds the row max), so the hot loop has no
// online-softmax rescaling at all.

#define N_NODES 8192
#define IN_FEAT 512
#define OUT_FEAT 64
#define LRELU_ALPHA 0.2f
#define NEG_BIG -9.0e15f

typedef __attribute__((ext_vector_type(16))) __bf16 v16bf;
typedef __attribute__((ext_vector_type(8)))  float  v8f;
typedef __attribute__((ext_vector_type(4)))  int    v4i;   // native clang vector
typedef __bf16 bf16_t;

// float -> bf16 bits, round-to-nearest-even (used off the critical path)
static __device__ __forceinline__ unsigned short f2bf(float f) {
  unsigned u = __float_as_uint(f);
  u += 0x7FFFu + ((u >> 16) & 1u);
  return (unsigned short)(u >> 16);
}

// pack two floats to packed bf16 (truncation) in ONE v_perm_b32
static __device__ __forceinline__ unsigned pack_bf16_pair(float lo, float hi) {
  // result bytes: [lo.b2, lo.b3, hi.b2, hi.b3]  (sel 0-3 -> src1, 4-7 -> src0)
  return __builtin_amdgcn_perm(__float_as_uint(hi), __float_as_uint(lo), 0x07060302u);
}

union V16BF  { v16bf v; unsigned short u[16]; };
union V16BFU { v16bf v; unsigned x[8]; };

// ---------------------------------------------------------------------------
// Pass A: Wh = h @ W via bf16 WMMA. One wave per 16x16 output tile.
// Writes Wh fp32 [N,64] and WhT bf16 column-major [64][N] (B-operand friendly:
// per output column, K is contiguous -> one 32B load per lane per K-tile).
// ---------------------------------------------------------------------------
__global__ __launch_bounds__(128) void gat_wh_wmma(
    const float* __restrict__ h, const float* __restrict__ W,
    float* __restrict__ wh, unsigned short* __restrict__ whT)
{
  const int lane  = threadIdx.x & 31;
  const int wid   = blockIdx.x * 4 + (threadIdx.x >> 5);   // 0..2047
  const int mtile = wid >> 2;                              // 0..511
  const int ntile = wid & 3;                               // 0..3
  const int i0    = mtile * 16;
  const int n0    = ntile * 16;
  const int half  = lane >> 4;
  const int mrow  = lane & 15;        // A: lane = row
  const int ncol  = n0 + (lane & 15); // B: lane = column
  const int koffA = half * 8;         // A lanes 0-15: K 0-7 & 16-23; 16-31: 8-15 & 24-31
  const int koffB = half * 16;

  v8f c;
  #pragma unroll
  for (int r = 0; r < 8; ++r) c[r] = 0.0f;

  const float* hrow = h + (size_t)(i0 + mrow) * IN_FEAT;

  for (int k0 = 0; k0 < IN_FEAT; k0 += 32) {
    float4 f0 = *(const float4*)(hrow + k0 + koffA);
    float4 f1 = *(const float4*)(hrow + k0 + koffA + 4);
    float4 f2 = *(const float4*)(hrow + k0 + koffA + 16);
    float4 f3 = *(const float4*)(hrow + k0 + koffA + 20);
    V16BF A;
    A.u[0]=f2bf(f0.x); A.u[1]=f2bf(f0.y); A.u[2]=f2bf(f0.z); A.u[3]=f2bf(f0.w);
    A.u[4]=f2bf(f1.x); A.u[5]=f2bf(f1.y); A.u[6]=f2bf(f1.z); A.u[7]=f2bf(f1.w);
    A.u[8]=f2bf(f2.x); A.u[9]=f2bf(f2.y); A.u[10]=f2bf(f2.z); A.u[11]=f2bf(f2.w);
    A.u[12]=f2bf(f3.x); A.u[13]=f2bf(f3.y); A.u[14]=f2bf(f3.z); A.u[15]=f2bf(f3.w);

    V16BF B;   // W is [512,64] row-major; column ncol, consecutive K (stride 64)
    #pragma unroll
    for (int t = 0; t < 16; ++t)
      B.u[t] = f2bf(W[(size_t)(k0 + koffA + (t & 7) + ((t >> 3) << 4)) * OUT_FEAT + ncol]);

    c = __builtin_amdgcn_wmma_f32_16x16x32_bf16(false, A.v, false, B.v,
                                                (short)0, c, false, false);
  }

  #pragma unroll
  for (int r = 0; r < 8; ++r) {
    const int row = i0 + r + 8 * half;        // C layout: VGPR r, lane half
    const int col = n0 + (lane & 15);
    const float v = c[r];
    wh[(size_t)row * OUT_FEAT + col] = v;
    whT[(size_t)col * N_NODES + row] = f2bf(v);
  }
}

// ---------------------------------------------------------------------------
// Pass B1: wh1[i] = Wh[i,:] . a[:64],  wh2[i] = Wh[i,:] . a[64:128]
// ---------------------------------------------------------------------------
__global__ __launch_bounds__(256) void gat_attvec(
    const float* __restrict__ wh, const float* __restrict__ a,
    float* __restrict__ wh1, float* __restrict__ wh2)
{
  const int i = blockIdx.x * 256 + threadIdx.x;
  if (i >= N_NODES) return;
  const float* r = wh + (size_t)i * OUT_FEAT;
  float s1 = 0.f, s2 = 0.f;
  #pragma unroll
  for (int t = 0; t < OUT_FEAT; t += 4) {
    float4 v  = *(const float4*)(r + t);
    float4 a1 = *(const float4*)(a + t);
    float4 a2 = *(const float4*)(a + OUT_FEAT + t);
    s1 += v.x*a1.x + v.y*a1.y + v.z*a1.z + v.w*a1.w;
    s2 += v.x*a2.x + v.y*a2.y + v.z*a2.z + v.w*a2.w;
  }
  wh1[i] = s1; wh2[i] = s2;
}

// ---------------------------------------------------------------------------
// Pass B2: global max of wh2 (single block, LDS tree reduction)
// ---------------------------------------------------------------------------
__global__ __launch_bounds__(256) void gat_wh2max(
    const float* __restrict__ wh2, float* __restrict__ out_max)
{
  __shared__ float red[256];
  float m = -3.0e38f;
  for (int i = threadIdx.x; i < N_NODES; i += 256) m = fmaxf(m, wh2[i]);
  red[threadIdx.x] = m;
  __syncthreads();
  #pragma unroll
  for (int s = 128; s > 0; s >>= 1) {
    if ((int)threadIdx.x < s) red[threadIdx.x] = fmaxf(red[threadIdx.x], red[threadIdx.x + s]);
    __syncthreads();
  }
  if (threadIdx.x == 0) *out_max = red[0];
}

// ---------------------------------------------------------------------------
// Pass C: fused masked-softmax + aggregation. One wave per 16-row block.
// Fixed per-row shift -> no online-softmax state, no accumulator rescaling.
// p = exp(masked ? lrelu(wh1+wh2)-C : NEG_BIG); exp(NEG_BIG)==0 masks for free.
// Probability tile packed to bf16 via v_perm and fed as WMMA A operand.
// ---------------------------------------------------------------------------
__global__ __launch_bounds__(128) void gat_attn_agg(
    const int* __restrict__ adj,
    const float* __restrict__ wh1v, const float* __restrict__ wh2v,
    const float* __restrict__ wh2max,
    const bf16_t* __restrict__ whT, float* __restrict__ out)
{
  const int lane   = threadIdx.x & 31;
  const int rowBlk = blockIdx.x * 4 + (threadIdx.x >> 5);  // 0..511
  const int i0     = rowBlk * 16;
  const int half   = lane >> 4;
  const int mrow   = lane & 15;
  const int koffA  = half * 8;
  const int koffB  = half * 16;
  const int i      = i0 + mrow;

  const float wh1_i = wh1v[i];
  const float q     = wh1_i + *wh2max;
  const float C     = fmaxf(q, LRELU_ALPHA * q);       // lrelu(wh1_i + max wh2)
  // score - C = max(wv + b1, 0.2*wv + b2)
  const float b1    = wh1_i - C;
  const float b2    = LRELU_ALPHA * wh1_i - C;

  const int* adjRow = adj + (size_t)i * N_NODES;
  const bf16_t* bp0 = whT + (size_t)(lane & 15) * N_NODES + koffB;

  float l_run = 0.0f;
  v8f acc0, acc1, acc2, acc3;
  #pragma unroll
  for (int r = 0; r < 8; ++r) { acc0[r]=0.f; acc1[r]=0.f; acc2[r]=0.f; acc3[r]=0.f; }

  for (int j0 = 0; j0 < N_NODES; j0 += 32) {
    __builtin_prefetch(adjRow + j0 + 1024, 0, 1);

    // adj: streamed once -> non-temporal (don't thrash L2 holding WhT)
    v4i a0 = __builtin_nontemporal_load((const v4i*)(adjRow + j0 + koffB));
    v4i a1 = __builtin_nontemporal_load((const v4i*)(adjRow + j0 + koffB + 4));
    v4i a2 = __builtin_nontemporal_load((const v4i*)(adjRow + j0 + koffB + 8));
    v4i a3 = __builtin_nontemporal_load((const v4i*)(adjRow + j0 + koffB + 12));
    float4 w0 = *(const float4*)(wh2v + j0 + koffB);
    float4 w1 = *(const float4*)(wh2v + j0 + koffB + 4);
    float4 w2 = *(const float4*)(wh2v + j0 + koffB + 8);
    float4 w3 = *(const float4*)(wh2v + j0 + koffB + 12);

    int msk[16]; float wv[16];
    msk[0]=a0.x; msk[1]=a0.y; msk[2]=a0.z; msk[3]=a0.w;
    msk[4]=a1.x; msk[5]=a1.y; msk[6]=a1.z; msk[7]=a1.w;
    msk[8]=a2.x; msk[9]=a2.y; msk[10]=a2.z; msk[11]=a2.w;
    msk[12]=a3.x; msk[13]=a3.y; msk[14]=a3.z; msk[15]=a3.w;
    wv[0]=w0.x; wv[1]=w0.y; wv[2]=w0.z; wv[3]=w0.w;
    wv[4]=w1.x; wv[5]=w1.y; wv[6]=w1.z; wv[7]=w1.w;
    wv[8]=w2.x; wv[9]=w2.y; wv[10]=w2.z; wv[11]=w2.w;
    wv[12]=w3.x; wv[13]=w3.y; wv[14]=w3.z; wv[15]=w3.w;

    float pp[16];
    #pragma unroll
    for (int t = 0; t < 16; ++t) {
      float sv = fmaxf(wv[t] + b1, fmaf(LRELU_ALPHA, wv[t], b2));  // lrelu - C
      sv = msk[t] ? sv : NEG_BIG;
      float p = __expf(sv);            // exp(NEG_BIG) == 0 -> masked
      l_run += p;
      pp[t] = p;
    }

    V16BFU A;
    #pragma unroll
    for (int k = 0; k < 8; ++k)
      A.x[k] = pack_bf16_pair(pp[2*k], pp[2*k + 1]);

    // B operand: WhT column-major -> contiguous 32B per lane (L2-resident)
    const bf16_t* bp = bp0 + j0;
    v16bf B0 = *(const v16bf*)(bp);
    v16bf B1 = *(const v16bf*)(bp + (size_t)16 * N_NODES);
    v16bf B2 = *(const v16bf*)(bp + (size_t)32 * N_NODES);
    v16bf B3 = *(const v16bf*)(bp + (size_t)48 * N_NODES);

    acc0 = __builtin_amdgcn_wmma_f32_16x16x32_bf16(false, A.v, false, B0, (short)0, acc0, false, false);
    acc1 = __builtin_amdgcn_wmma_f32_16x16x32_bf16(false, A.v, false, B1, (short)0, acc1, false, false);
    acc2 = __builtin_amdgcn_wmma_f32_16x16x32_bf16(false, A.v, false, B2, (short)0, acc2, false, false);
    acc3 = __builtin_amdgcn_wmma_f32_16x16x32_bf16(false, A.v, false, B3, (short)0, acc3, false, false);
  }

  // combine K-halves of each row, then divide + ELU + store
  const float l_tot = l_run + __shfl_xor(l_run, 16, 32);
  const float linv  = 1.0f / fmaxf(l_tot, 1e-30f);
  #pragma unroll
  for (int r = 0; r < 8; ++r) {
    float lr = __shfl(linv, r + 8 * half, 32);   // row of VGPR r is (r + 8*half)
    const int row = i0 + r + 8 * half;
    float* o = out + (size_t)row * OUT_FEAT + (lane & 15);
    float v0 = acc0[r] * lr; v0 = (v0 > 0.f) ? v0 : (__expf(v0) - 1.f);  // ELU
    float v1 = acc1[r] * lr; v1 = (v1 > 0.f) ? v1 : (__expf(v1) - 1.f);
    float v2 = acc2[r] * lr; v2 = (v2 > 0.f) ? v2 : (__expf(v2) - 1.f);
    float v3 = acc3[r] * lr; v3 = (v3 > 0.f) ? v3 : (__expf(v3) - 1.f);
    o[0] = v0; o[16] = v1; o[32] = v2; o[48] = v3;
  }
}

// ---------------------------------------------------------------------------
extern "C" void kernel_launch(void* const* d_in, const int* in_sizes, int n_in,
                              void* d_out, int out_size, void* d_ws, size_t ws_size,
                              hipStream_t stream)
{
  const float* h   = (const float*)d_in[0];   // [8192,512] f32
  const int*   adj = (const int*)  d_in[1];   // [8192,8192] i32
  const float* W   = (const float*)d_in[2];   // [512,64] f32
  const float* a   = (const float*)d_in[3];   // [128,1] f32
  float* out = (float*)d_out;                 // [8192,64] f32

  // Workspace layout (~3.07 MB):
  //   [0, 2MB)          Wh  fp32 [8192][64]
  //   [2MB, 3MB)        WhT bf16 [64][8192] (column-major, WMMA B operand)
  //   [3MB, +32KB)      wh1[8192] f32
  //   [3MB+32KB, ...)   wh2[8192] f32
  //   [3MB+64KB]        wh2max (1 float)
  char* ws = (char*)d_ws;
  float*          wh   = (float*)ws;
  unsigned short* whT  = (unsigned short*)(ws + (size_t)N_NODES * OUT_FEAT * 4);
  float*          wh1  = (float*)(ws + (size_t)3 * 1024 * 1024);
  float*          wh2  = wh1 + N_NODES;
  float*          w2mx = wh2 + N_NODES;

  gat_wh_wmma<<<512, 128, 0, stream>>>(h, W, wh, whT);
  gat_attvec<<<N_NODES / 256, 256, 0, stream>>>(wh, a, wh1, wh2);
  gat_wh2max<<<1, 256, 0, stream>>>(wh2, w2mx);
  gat_attn_agg<<<128, 128, 0, stream>>>(adj, wh1, wh2, w2mx,
                                        (const bf16_t*)whT, out);
}
